// SimplicialAttentionLayer2_7172595384926
// MI455X (gfx1250) — matrix operand order repaired and checked by reading, not compiled, not run on hardware
//
#include <hip/hip_runtime.h>
#include <cstdint>
#include <cstddef>

// ---------------------------------------------------------------------------
// Types for CDNA5 WMMA (gfx1250, wave32)
// ---------------------------------------------------------------------------
typedef __attribute__((ext_vector_type(16))) __bf16   v16bf;
typedef __attribute__((ext_vector_type(8)))  float    v8f;
typedef __attribute__((ext_vector_type(8)))  unsigned v8u;

#define NW    2            // waves per block
#define BD    (NW * 32)    // 64 threads per block
#define DHEAD 256          // embedding dim
#define KP    (DHEAD / 2)  // 128 uint (bf16-pair) per row
#define SPLIT 8            // source-dimension chunks (flash-decoding split)

static __device__ __forceinline__ v8f vzero() {
  v8f z = {0.f, 0.f, 0.f, 0.f, 0.f, 0.f, 0.f, 0.f};
  return z;
}

static __device__ __forceinline__ v8f wmma_bf16(v8u a, v8u b, v8f c) {
  // D = A(16x32 bf16) * B(32x16 bf16) + C(16x16 f32)
  return __builtin_amdgcn_wmma_f32_16x16x32_bf16(
      false, __builtin_bit_cast(v16bf, a),
      false, __builtin_bit_cast(v16bf, b),
      (short)0, c, false, false);
}

static __device__ __forceinline__ unsigned short f32_to_bf16r(float f) {
  unsigned u = __builtin_bit_cast(unsigned, f);
  return (unsigned short)((u + 0x7FFFu + ((u >> 16) & 1u)) >> 16);
}

// ---------------------------------------------------------------------------
// fp32 -> bf16 conversion (RNE)
// ---------------------------------------------------------------------------
__global__ void cvt_bf16_kernel(const float* __restrict__ in,
                                unsigned short* __restrict__ out, int n) {
  int i = blockIdx.x * blockDim.x + threadIdx.x;
  if (i < n) out[i] = f32_to_bf16r(in[i]);
}

__global__ void zero_kernel(float* __restrict__ p, int n) {
  int i = blockIdx.x * blockDim.x + threadIdx.x;
  if (i < n) p[i] = 0.f;
}

// ---------------------------------------------------------------------------
// S = Q * K^T for a 16(query) x 32(source) tile, K-reduction over DHEAD=256.
// A-frag (16-bit A 16x32): m = lane&15; uint r packs
//   k = kk*32 + (lane>=16 ? 16:0) + 2r .. +1
// B-frag (16-bit B 32x16): n = lane&15; uint r packs k = (lane>=16?16:0)+2r..+1
// ---------------------------------------------------------------------------
static __device__ __forceinline__ void compute_S(
    const unsigned* ldsQ, const unsigned* ldsK,
    int qrow0, int lane, v8f& c0, v8f& c1)
{
  const int m  = lane & 15;
  const int hi = (lane >> 4) & 1;
  c0 = vzero();
  c1 = vzero();
#pragma unroll
  for (int kk = 0; kk < 8; ++kk) {
    int qo = (qrow0 + m) * KP + kk * 16 + hi * 8;
    v8u a = { ldsQ[qo + 0], ldsQ[qo + 1], ldsQ[qo + 2],  ldsQ[qo + 3],
              ldsQ[qo + 4], ldsQ[qo + 5], ldsQ[qo + 6], ldsQ[qo + 7] };
    int b0 = m * KP + kk * 16 + hi * 8;          // src rows 0..15
    v8u bl = { ldsK[b0 + 0], ldsK[b0 + 1], ldsK[b0 + 2], ldsK[b0 + 3],
               ldsK[b0 + 4], ldsK[b0 + 5], ldsK[b0 + 6], ldsK[b0 + 7] };
    int b1 = b0 + 16 * KP;                        // src rows 16..31
    v8u bh = { ldsK[b1 + 0], ldsK[b1 + 1], ldsK[b1 + 2], ldsK[b1 + 3],
               ldsK[b1 + 4], ldsK[b1 + 5], ldsK[b1 + 6], ldsK[b1 + 7] };
    c0 = wmma_bf16(a, bl, c0);
    c1 = wmma_bf16(a, bh, c1);
  }
}

// ---------------------------------------------------------------------------
// Pass 1 (per source chunk): partial row-max / exp-sum of s = Q K^T / 16.
// K tile is staged global->LDS with the gfx1250 async engine
// (global_load_async_to_lds_b128 + s_wait_asynccnt), no VGPR round-trip.
// Single shared allocation puts the K tile at LDS offset 0 so the async
// destination offset is exact.
// ---------------------------------------------------------------------------
__global__ __launch_bounds__(BD)
void attn_pass1_kernel(const unsigned* __restrict__ Qb,
                       const unsigned* __restrict__ Sb,
                       float* __restrict__ Mpart,
                       float* __restrict__ Zpart,
                       int Cn)
{
  __shared__ unsigned shm[32 * KP + NW * 16 * KP]; // [0,16KB): K tile, then Q
  unsigned* ldsK = shm;
  unsigned* ldsQ = shm + 32 * KP;

  const int tid  = threadIdx.x;
  const int lane = tid & 31;
  const int wave = tid >> 5;
  const int m    = lane & 15;
  const int hi   = (lane >> 4) & 1;
  const int chunk  = blockIdx.y;
  const int qblock = blockIdx.x * (NW * 16);

  for (int i = tid; i < NW * 16 * KP; i += BD)
    ldsQ[i] = Qb[(size_t)qblock * KP + i];

  const int qrow0 = wave * 16;
  const int qg0   = qblock + qrow0;

  float M[8], Z[8];
#pragma unroll
  for (int r = 0; r < 8; ++r) { M[r] = -1e30f; Z[r] = 0.f; }

  const float sc = 0.0625f;     // 1/sqrt(256)
  const int nt = Cn >> 5;
  const size_t chbase = (size_t)chunk * Cn * KP;

  for (int t = 0; t < nt; ++t) {
    // async copy of the 16 KB K tile: 1024 x b128, LDS offsets 0..16KB
    const unsigned long gk =
        (unsigned long)(Sb + chbase + (size_t)t * 32 * KP);
    for (int j = tid; j < 1024; j += BD) {
      unsigned       ldst = (unsigned)(j * 16);          // LDS byte offset
      unsigned long  ga   = gk + (unsigned long)j * 16;  // global byte addr
      asm volatile("global_load_async_to_lds_b128 %0, %1, off"
                   :: "v"(ldst), "v"(ga) : "memory");
    }
    asm volatile("s_wait_asynccnt 0x0" ::: "memory");
    __syncthreads();

    v8f c0, c1;
    compute_S(ldsQ, ldsK, qrow0, lane, c0, c1);

#pragma unroll
    for (int r = 0; r < 8; ++r) {
      float s0 = c0[r] * sc, s1 = c1[r] * sc;
      float mx = fmaxf(s0, s1);
#pragma unroll
      for (int i = 1; i < 16; i <<= 1) mx = fmaxf(mx, __shfl_xor(mx, i, 16));
      float nm = fmaxf(M[r], mx);
      float e  = __expf(s0 - nm) + __expf(s1 - nm);
#pragma unroll
      for (int i = 1; i < 16; i <<= 1) e += __shfl_xor(e, i, 16);
      Z[r] = Z[r] * __expf(M[r] - nm) + e;
      M[r] = nm;
    }
    __syncthreads();
  }

  if (m == 0) {
#pragma unroll
    for (int r = 0; r < 8; ++r) {
      int qg = qg0 + r + hi * 8;
      Mpart[(size_t)qg * SPLIT + chunk] = M[r];
      Zpart[(size_t)qg * SPLIT + chunk] = Z[r];
    }
  }
}

// ---------------------------------------------------------------------------
// Combine per-chunk stats into whole-row max / denominator.
// ---------------------------------------------------------------------------
__global__ void reduce_stats_kernel(const float* __restrict__ Mp,
                                    const float* __restrict__ Zp,
                                    float* __restrict__ Mrow,
                                    float* __restrict__ Zrow, int n) {
  int i = blockIdx.x * blockDim.x + threadIdx.x;
  if (i >= n) return;
  float M = -1e30f;
#pragma unroll
  for (int c = 0; c < SPLIT; ++c) M = fmaxf(M, Mp[(size_t)i * SPLIT + c]);
  float Z = 0.f;
#pragma unroll
  for (int c = 0; c < SPLIT; ++c)
    Z += Zp[(size_t)i * SPLIT + c] * __expf(Mp[(size_t)i * SPLIT + c] - M);
  Mrow[i] = M;
  Zrow[i] = Z;
}

// ---------------------------------------------------------------------------
// Pass 2 (per source chunk): p = exp(s-M)/Z, w = mask ? exp(p) : 1 (= exp(p*m)),
// accumulate partial sum(w) and partial w@V; combine across chunks with
// global f32 atomics. Mask tiles staged through LDS with coalesced loads in
// BOTH layouts (fixes the strided transposed-mask gathers of term 2).
// ---------------------------------------------------------------------------
__global__ __launch_bounds__(BD)
void attn_pass2_kernel(const unsigned* __restrict__ Qb,
                       const unsigned* __restrict__ Sb,
                       const int*      __restrict__ mask,
                       const float*    __restrict__ Mrow,
                       const float*    __restrict__ Zrow,
                       float* __restrict__ accbuf,
                       float* __restrict__ z2buf,
                       int Cn, int mask_ld, int transposed)
{
  __shared__ unsigned ldsQ[NW * 16 * KP];  // 16 KB
  __shared__ unsigned ldsK[32 * KP];       // 16 KB row-major (for S)
  __shared__ unsigned ldsVt[DHEAD * 16];   // 16 KB transposed V [col][kpair]
  __shared__ unsigned ldsP[NW * 16 * 16];  //  2 KB per-wave P (16x32 bf16)
  __shared__ int      ldsM[32 * 32];       //  4 KB mask tile

  const int tid  = threadIdx.x;
  const int lane = tid & 31;
  const int wave = tid >> 5;
  const int m    = lane & 15;
  const int hi   = (lane >> 4) & 1;
  const int chunk  = blockIdx.y;
  const int sbase  = chunk * Cn;
  const int qblock = blockIdx.x * (NW * 16);

  for (int i = tid; i < NW * 16 * KP; i += BD)
    ldsQ[i] = Qb[(size_t)qblock * KP + i];
  __syncthreads();

  const int qrow0 = wave * 16;
  const int qg0   = qblock + qrow0;

  float Mr[8], iZ[8], Z2[8];
#pragma unroll
  for (int r = 0; r < 8; ++r) {
    int qg = qg0 + r + hi * 8;
    Mr[r] = Mrow[qg];
    iZ[r] = 1.0f / Zrow[qg];
    Z2[r] = 0.f;
  }

  v8f acc[16];
#pragma unroll
  for (int ct = 0; ct < 16; ++ct) acc[ct] = vzero();

  const float sc = 0.0625f;
  const int nt = Cn >> 5;

  for (int t = 0; t < nt; ++t) {
    const int sb0 = sbase + t * 32;
    const unsigned* gk = Sb + (size_t)sb0 * KP;
    for (int i = tid; i < 32 * KP; i += BD) {
      unsigned u = gk[i];
      ldsK[i] = u;
      int r  = i >> 7;            // source row 0..31
      int cp = i & (KP - 1);      // column pair
      unsigned short* vt = (unsigned short*)ldsVt;
      vt[(2 * cp + 0) * 32 + r] = (unsigned short)(u & 0xFFFFu);
      vt[(2 * cp + 1) * 32 + r] = (unsigned short)(u >> 16);
    }
    // coalesced mask tile (32 src x 32 q)
    for (int i = tid; i < 1024; i += BD) {
      if (transposed) {
        int sl = i >> 5, ql = i & 31;
        ldsM[sl * 32 + ql] = mask[(size_t)(sb0 + sl) * mask_ld + (qblock + ql)];
      } else {
        int ql = i >> 5, sl = i & 31;
        ldsM[ql * 32 + sl] = mask[(size_t)(qblock + ql) * mask_ld + (sb0 + sl)];
      }
    }
    if (t + 1 < nt)
      __builtin_prefetch(gk + 32 * KP + tid, 0, 0);   // global_prefetch_b8
    __syncthreads();

    v8f c0, c1;
    compute_S(ldsQ, ldsK, qrow0, lane, c0, c1);

    unsigned short* p16 = (unsigned short*)ldsP + wave * 16 * 32;
#pragma unroll
    for (int r = 0; r < 8; ++r) {
      float p0 = __expf(c0[r] * sc - Mr[r]) * iZ[r];
      float p1 = __expf(c1[r] * sc - Mr[r]) * iZ[r];
      int ql = qrow0 + r + hi * 8;      // query-local 0..31
      int m0 = transposed ? ldsM[m * 32 + ql]        : ldsM[ql * 32 + m];
      int m1 = transposed ? ldsM[(m + 16) * 32 + ql] : ldsM[ql * 32 + m + 16];
      float w0 = m0 ? __expf(p0) : 1.0f;   // exp(p*mask); exp(0)=1
      float w1 = m1 ? __expf(p1) : 1.0f;
      float e = w0 + w1;
#pragma unroll
      for (int i = 1; i < 16; i <<= 1) e += __shfl_xor(e, i, 16);
      Z2[r] += e;
      int mr = r + hi * 8;
      p16[mr * 32 + m]      = f32_to_bf16r(w0);
      p16[mr * 32 + 16 + m] = f32_to_bf16r(w1);
    }

    // P(16x32) @ V(32x256)
    const unsigned* ldsPw = ldsP + wave * 16 * 16;
    int po = m * 16 + hi * 8;
    v8u ap = { ldsPw[po + 0], ldsPw[po + 1], ldsPw[po + 2],  ldsPw[po + 3],
               ldsPw[po + 4], ldsPw[po + 5], ldsPw[po + 6], ldsPw[po + 7] };
#pragma unroll
    for (int ct = 0; ct < 16; ++ct) {
      int vb = (ct * 16 + m) * 16 + hi * 8;
      v8u bv = { ldsVt[vb + 0], ldsVt[vb + 1], ldsVt[vb + 2], ldsVt[vb + 3],
                 ldsVt[vb + 4], ldsVt[vb + 5], ldsVt[vb + 6], ldsVt[vb + 7] };
      acc[ct] = wmma_bf16(ap, bv, acc[ct]);
    }
    __syncthreads();
  }

  // combine partials across chunks
  if (m == 0) {
#pragma unroll
    for (int r = 0; r < 8; ++r)
      atomicAdd(&z2buf[qg0 + r + hi * 8], Z2[r]);
  }
#pragma unroll
  for (int ct = 0; ct < 16; ++ct) {
#pragma unroll
    for (int r = 0; r < 8; ++r) {
      int qg = qg0 + r + hi * 8;
      atomicAdd(&accbuf[(size_t)qg * DHEAD + ct * 16 + m], acc[ct][r]);
    }
  }
}

// ---------------------------------------------------------------------------
// out (+)= (acc / sum(w)) / 3
// ---------------------------------------------------------------------------
__global__ void finalize_kernel(const float* __restrict__ accbuf,
                                const float* __restrict__ z2buf,
                                float* __restrict__ out, int n, int accumulate) {
  int i = blockIdx.x * blockDim.x + threadIdx.x;
  if (i >= n) return;
  float v = accbuf[i] * ((1.0f / 3.0f) / z2buf[i >> 8]);
  if (accumulate) out[i] += v;
  else            out[i]  = v;
}

// ---------------------------------------------------------------------------
// Launch. Inputs: 0:L[8192,8192] 1:H[8192,256] 2:B_low[4096,8192]
//                 3:H_low[4096,256] 4:B_high[8192,16384] 5:H_high[16384,256]
// ---------------------------------------------------------------------------
extern "C" void kernel_launch(void* const* d_in, const int* in_sizes, int n_in,
                              void* d_out, int out_size, void* d_ws, size_t ws_size,
                              hipStream_t stream) {
  (void)in_sizes; (void)n_in; (void)out_size; (void)ws_size;

  const int*   L  = (const int*)  d_in[0];
  const float* H  = (const float*)d_in[1];
  const int*   Bl = (const int*)  d_in[2];
  const float* Hl = (const float*)d_in[3];
  const int*   Bh = (const int*)  d_in[4];
  const float* Hh = (const float*)d_in[5];
  float* out = (float*)d_out;

  const int N = 8192, Nlo = 4096, Nhi = 16384, D = DHEAD;

  // workspace layout
  unsigned short* Hb   = (unsigned short*)d_ws;
  unsigned short* Hlb  = Hb  + (size_t)N   * D;
  unsigned short* Hhb  = Hlb + (size_t)Nlo * D;
  float* accbuf = (float*)(Hhb + (size_t)Nhi * D);
  float* z2buf  = accbuf + (size_t)N * D;
  float* Mrow   = z2buf + N;
  float* Zrow   = Mrow + N;
  float* Mpart  = Zrow + N;
  float* Zpart  = Mpart + (size_t)N * SPLIT;

  const int nH = N * D, nHl = Nlo * D, nHh = Nhi * D;
  cvt_bf16_kernel<<<(nH  + 255) / 256, 256, 0, stream>>>(H,  Hb,  nH);
  cvt_bf16_kernel<<<(nHl + 255) / 256, 256, 0, stream>>>(Hl, Hlb, nHl);
  cvt_bf16_kernel<<<(nHh + 255) / 256, 256, 0, stream>>>(Hh, Hhb, nHh);

  const unsigned short* srcb[3] = { Hb, Hlb, Hhb };
  const int*  msk[3]  = { L, Bl, Bh };
  const int   nsrc[3] = { N, Nlo, Nhi };
  const int   mld[3]  = { N, N, Nhi };    // B_low stored [N_low, N] -> transposed
  const int   trs[3]  = { 0, 1, 0 };

  dim3 block(BD);
  dim3 gattn(N / (NW * 16), SPLIT);

  for (int term = 0; term < 3; ++term) {
    const int Cn = nsrc[term] / SPLIT;
    zero_kernel<<<(N * D + 255) / 256, 256, 0, stream>>>(accbuf, N * D);
    zero_kernel<<<(N + 255) / 256, 256, 0, stream>>>(z2buf, N);
    attn_pass1_kernel<<<gattn, block, 0, stream>>>(
        (const unsigned*)Hb, (const unsigned*)srcb[term], Mpart, Zpart, Cn);
    reduce_stats_kernel<<<(N + 255) / 256, 256, 0, stream>>>(
        Mpart, Zpart, Mrow, Zrow, N);
    attn_pass2_kernel<<<gattn, block, 0, stream>>>(
        (const unsigned*)Hb, (const unsigned*)srcb[term], msk[term],
        Mrow, Zrow, accbuf, z2buf, Cn, mld[term], trs[term]);
    finalize_kernel<<<(N * D + 255) / 256, 256, 0, stream>>>(
        accbuf, z2buf, out, N * D, term > 0);
  }
}
